// MoF_13640816132304
// MI455X (gfx1250) — matrix operand order repaired and checked by reading, not compiled, not run on hardware
//
#include <hip/hip_runtime.h>

// ---------------- CDNA5 WMMA types ----------------
typedef __attribute__((ext_vector_type(16))) __bf16 v16bf;
typedef __attribute__((ext_vector_type(8)))  __bf16 v8bf;
typedef __attribute__((ext_vector_type(8)))  float  v8f;
typedef __attribute__((ext_vector_type(4)))  int    v4i;

union AFrag { v16bf full; v8bf half[2]; };

// Problem sizes (match reference: B=2, L=2048, D=2048, E=8, Dg=256, k=2)
constexpr int T  = 4096;   // B*L tokens
constexpr int D  = 2048;
constexpr int E  = 8;
constexpr int DG = 256;

// -------- async global->LDS 16-byte copy (CDNA5) --------
__device__ __forceinline__ void async_copy16(const unsigned short* gsrc,
                                             unsigned short* ldst) {
#if __has_builtin(__builtin_amdgcn_global_load_async_to_lds_b128)
  __builtin_amdgcn_global_load_async_to_lds_b128((v4i*)gsrc, (v4i*)ldst, 0, 0);
#else
  unsigned lo = (unsigned)(size_t)ldst;   // low 32 bits of flat LDS addr = LDS offset
  asm volatile("global_load_async_to_lds_b128 %0, %1, off"
               :: "v"(lo), "v"(gsrc) : "memory");
#endif
}

__device__ __forceinline__ void wait_async0() {
#if __has_builtin(__builtin_amdgcn_s_wait_asynccnt)
  __builtin_amdgcn_s_wait_asynccnt(0);
#else
  asm volatile("s_wait_asynccnt 0" ::: "memory");
#endif
}

__device__ __forceinline__ unsigned short f2bf(float f) {
  unsigned int u = __float_as_uint(f);
  u += 0x7FFFu + ((u >> 16) & 1u);       // round-to-nearest-even
  return (unsigned short)(u >> 16);
}

// ---------------- 1) zero accumulators ----------------
__global__ void moe_init_zero(float* __restrict__ out, float* __restrict__ down,
                              int* __restrict__ counts) {
  size_t i = (size_t)blockIdx.x * blockDim.x + threadIdx.x;
  size_t stride = (size_t)gridDim.x * blockDim.x;
  for (size_t j = i; j < (size_t)T * D;  j += stride) out[j]  = 0.f;
  for (size_t j = i; j < (size_t)T * DG; j += stride) down[j] = 0.f;
  if (i < (size_t)E) counts[i] = 0;
}

// ---------------- 2) f32 -> bf16 conversion ----------------
__global__ void moe_cvt_bf16(const float* __restrict__ src,
                             unsigned short* __restrict__ dst, size_t n) {
  size_t i = (size_t)blockIdx.x * blockDim.x + threadIdx.x;
  size_t stride = (size_t)gridDim.x * blockDim.x;
  for (size_t j = i; j < n; j += stride) dst[j] = f2bf(src[j]);
}

// ---------------- 3) gate: scores + top-2 + routing + x->bf16 ----------------
__global__ void moe_gate(const float* __restrict__ x, const float* __restrict__ Wg,
                         unsigned short* __restrict__ xb,
                         int* __restrict__ counts, int* __restrict__ lists,
                         float* __restrict__ gatew) {
  const int t   = blockIdx.x;       // one block per token
  const int tid = threadIdx.x;      // 256 threads
  const float* xr = x + (size_t)t * D;

  float acc[E];
#pragma unroll
  for (int e = 0; e < E; ++e) acc[e] = 0.f;

  for (int d = tid; d < D; d += 256) {
    float xv = xr[d];
    xb[(size_t)t * D + d] = f2bf(xv);
#pragma unroll
    for (int e = 0; e < E; ++e) acc[e] += xv * Wg[e * D + d];
  }

  __shared__ float red[E][256];
#pragma unroll
  for (int e = 0; e < E; ++e) red[e][tid] = acc[e];
  __syncthreads();
  for (int s = 128; s > 0; s >>= 1) {
    if (tid < s) {
#pragma unroll
      for (int e = 0; e < E; ++e) red[e][tid] += red[e][tid + s];
    }
    __syncthreads();
  }

  if (tid == 0) {
    float sv[E];
#pragma unroll
    for (int e = 0; e < E; ++e) sv[e] = 1.f / (1.f + __expf(-red[e][0]));
    int i0 = 0; float g0 = sv[0];
#pragma unroll
    for (int e = 1; e < E; ++e) if (sv[e] > g0) { g0 = sv[e]; i0 = e; }
    int i1 = (i0 == 0) ? 1 : 0; float g1 = sv[i1];
#pragma unroll
    for (int e = 0; e < E; ++e) if (e != i0 && sv[e] > g1) { g1 = sv[e]; i1 = e; }

    int p0 = atomicAdd(&counts[i0], 1);
    lists[i0 * T + p0] = t; gatew[i0 * T + p0] = g0;
    int p1 = atomicAdd(&counts[i1], 1);
    lists[i1 * T + p1] = t; gatew[i1 * T + p1] = g1;
  }
}

// ---------------- 4) grouped GEMM: down += w * (x @ Wd[e]^T) ----------------
// grid: (T/16, E), block: 128 (4 waves). Tile: 16 tokens x 256 (Dg), K over D.
// B K-panels (16 KB) + gathered A rows staged into LDS with async copies,
// double-buffered, overlapped with WMMA compute.
__global__ void moe_down_gemm(const unsigned short* __restrict__ xb,
                              const unsigned short* __restrict__ Wdb,
                              const int* __restrict__ lists,
                              const float* __restrict__ gatew,
                              const int* __restrict__ counts,
                              float* __restrict__ down) {
  const int e    = blockIdx.y;
  const int tile = blockIdx.x;
  const int ne   = counts[e];
  if (tile * 16 >= ne) return;

  __shared__ int   s_tok[16];
  __shared__ float s_w[16];
  __shared__ __align__(32) unsigned short sB[2][DG * 32];  // 2 x 16 KB
  __shared__ __align__(32) unsigned short sA[2][16 * 32];  // 2 x  1 KB

  const int tid = threadIdx.x;
  if (tid < 16) {
    int idx = tile * 16 + tid;
    bool ok = idx < ne;
    s_tok[tid] = ok ? lists[e * T + idx] : 0;
    s_w[tid]   = ok ? gatew[e * T + idx] : 0.f;
  }
  __syncthreads();

  const unsigned short* Bbase = Wdb + (size_t)e * DG * D;

  auto stage = [&](int kk, int buf) {
    // B panel: 256 rows x 32 K -> 1024 x 16B chunks, 8 per thread
#pragma unroll
    for (int it = 0; it < 8; ++it) {
      int c   = tid + it * 128;
      int row = c >> 2, sub = c & 3;
      async_copy16(Bbase + (size_t)row * D + kk * 32 + sub * 8,
                   &sB[buf][row * 32 + sub * 8]);
    }
    // A panel: 16 gathered token rows x 32 K -> 64 x 16B chunks
    if (tid < 64) {
      int row = tid >> 2, sub = tid & 3;
      async_copy16(xb + (size_t)s_tok[row] * D + kk * 32 + sub * 8,
                   &sA[buf][row * 32 + sub * 8]);
    }
  };

  stage(0, 0);
  wait_async0();
  __syncthreads();

  const int wave = tid >> 5;
  const int lane = tid & 31;
  const int m    = lane & 15;
  const int h    = lane >> 4;

  v8f c[4] = {};
  for (int kk = 0; kk < D / 32; ++kk) {
    const int buf = kk & 1;
    if (kk + 1 < D / 32) stage(kk + 1, buf ^ 1);

    // Issue ALL fragment loads first (distinct registers), then the WMMAs,
    // so LDS latency pipelines against the matrix pipe instead of wait-0
    // before every WMMA.
    AFrag a;
    const v8bf* ap = (const v8bf*)&sA[buf][m * 32 + 16 * h];
    a.half[0] = ap[0];              // K = 16h .. 16h+7
    a.half[1] = ap[1];              // K = 16h+8 .. 16h+15

    v16bf b[4];
#pragma unroll
    for (int j = 0; j < 4; ++j) {
      const int row = (wave * 4 + j) * 16 + m;
      b[j] = *(const v16bf*)&sB[buf][row * 32 + 16 * h];
    }
#pragma unroll
    for (int j = 0; j < 4; ++j) {
      c[j] = __builtin_amdgcn_wmma_f32_16x16x32_bf16(
          false, a.full, false, b[j], (short)0, c[j], false, false);
    }

    if (kk + 1 < D / 32) { wait_async0(); __syncthreads(); }
  }

#pragma unroll
  for (int j = 0; j < 4; ++j) {
    const int col = (wave * 4 + j) * 16 + m;
#pragma unroll
    for (int r = 0; r < 8; ++r) {
      const int   M  = r + 8 * h;
      const float wv = s_w[M];
      if (wv != 0.f)
        atomicAdd(&down[(size_t)s_tok[M] * DG + col], wv * c[j][r]);
    }
  }
}

// ---------------- 5) grouped GEMM: up += w * (down @ Wu[e]^T) ----------------
// grid: (T/16, E, 8). Each block: 16 tokens x 256 output cols, K = Dg = 256.
// Same LDS async double-buffer scheme as the down GEMM.
__global__ void moe_up_gemm(const unsigned short* __restrict__ downb,
                            const unsigned short* __restrict__ Wub,
                            const int* __restrict__ lists,
                            const float* __restrict__ gatew,
                            const int* __restrict__ counts,
                            float* __restrict__ out) {
  const int e    = blockIdx.y;
  const int tile = blockIdx.x;
  const int ne   = counts[e];
  if (tile * 16 >= ne) return;

  const int rbase = blockIdx.z * 256;     // output-column base (panel of Wu rows)

  __shared__ int   s_tok[16];
  __shared__ float s_w[16];
  __shared__ __align__(32) unsigned short sB[2][256 * 32]; // 2 x 16 KB
  __shared__ __align__(32) unsigned short sA[2][16 * 32];  // 2 x  1 KB

  const int tid = threadIdx.x;
  if (tid < 16) {
    int idx = tile * 16 + tid;
    bool ok = idx < ne;
    s_tok[tid] = ok ? lists[e * T + idx] : 0;
    s_w[tid]   = ok ? gatew[e * T + idx] : 0.f;
  }
  __syncthreads();

  const unsigned short* Bbase = Wub + ((size_t)e * D + rbase) * DG;

  auto stage = [&](int kk, int buf) {
#pragma unroll
    for (int it = 0; it < 8; ++it) {
      int c   = tid + it * 128;
      int row = c >> 2, sub = c & 3;
      async_copy16(Bbase + (size_t)row * DG + kk * 32 + sub * 8,
                   &sB[buf][row * 32 + sub * 8]);
    }
    if (tid < 64) {
      int row = tid >> 2, sub = tid & 3;
      async_copy16(downb + (size_t)s_tok[row] * DG + kk * 32 + sub * 8,
                   &sA[buf][row * 32 + sub * 8]);
    }
  };

  stage(0, 0);
  wait_async0();
  __syncthreads();

  const int wave = tid >> 5;
  const int lane = tid & 31;
  const int m    = lane & 15;
  const int h    = lane >> 4;

  v8f c[4] = {};
  for (int kk = 0; kk < DG / 32; ++kk) {
    const int buf = kk & 1;
    if (kk + 1 < DG / 32) stage(kk + 1, buf ^ 1);

    AFrag a;
    const v8bf* ap = (const v8bf*)&sA[buf][m * 32 + 16 * h];
    a.half[0] = ap[0];
    a.half[1] = ap[1];

    v16bf b[4];
#pragma unroll
    for (int j = 0; j < 4; ++j) {
      const int row = (wave * 4 + j) * 16 + m;
      b[j] = *(const v16bf*)&sB[buf][row * 32 + 16 * h];
    }
#pragma unroll
    for (int j = 0; j < 4; ++j) {
      c[j] = __builtin_amdgcn_wmma_f32_16x16x32_bf16(
          false, a.full, false, b[j], (short)0, c[j], false, false);
    }

    if (kk + 1 < DG / 32) { wait_async0(); __syncthreads(); }
  }

#pragma unroll
  for (int j = 0; j < 4; ++j) {
    const int col = rbase + (wave * 4 + j) * 16 + m;
#pragma unroll
    for (int r = 0; r < 8; ++r) {
      const int   M  = r + 8 * h;
      const float wv = s_w[M];
      if (wv != 0.f)
        atomicAdd(&out[(size_t)s_tok[M] * D + col], wv * c[j][r]);
    }
  }
}

// ---------------- host launcher ----------------
extern "C" void kernel_launch(void* const* d_in, const int* in_sizes, int n_in,
                              void* d_out, int out_size, void* d_ws, size_t ws_size,
                              hipStream_t stream) {
  (void)in_sizes; (void)n_in; (void)out_size; (void)ws_size;

  const float* x  = (const float*)d_in[0];   // [T, D]
  const float* Wg = (const float*)d_in[1];   // [E, D]
  const float* Wd = (const float*)d_in[2];   // [E, Dg, D]
  const float* Wu = (const float*)d_in[3];   // [E, D, Dg]
  float* out = (float*)d_out;                // [T, D]

  // Workspace layout (all offsets 256B-aligned), total ~40.1 MB
  char* ws = (char*)d_ws;
  unsigned short* xb    = (unsigned short*)(ws + 0);           // T*D bf16
  unsigned short* Wdb   = (unsigned short*)(ws + 16777216);    // E*Dg*D bf16
  unsigned short* Wub   = (unsigned short*)(ws + 25165824);    // E*D*Dg bf16
  float*          down  = (float*)         (ws + 33554432);    // T*Dg f32
  unsigned short* downb = (unsigned short*)(ws + 37748736);    // T*Dg bf16
  int*            counts= (int*)           (ws + 39845888);    // E ints
  int*            lists = (int*)           (ws + 39846144);    // E*T ints
  float*          gatew = (float*)         (ws + 39977216);    // E*T floats

  moe_init_zero<<<4096, 256, 0, stream>>>(out, down, counts);

  moe_cvt_bf16<<<4096, 256, 0, stream>>>(Wd, Wdb, (size_t)E * DG * D);
  moe_cvt_bf16<<<4096, 256, 0, stream>>>(Wu, Wub, (size_t)E * D * DG);

  moe_gate<<<T, 256, 0, stream>>>(x, Wg, xb, counts, lists, gatew);

  moe_down_gemm<<<dim3(T / 16, E), 128, 0, stream>>>(xb, Wdb, lists, gatew, counts, down);

  moe_cvt_bf16<<<1024, 256, 0, stream>>>(down, downb, (size_t)T * DG);

  moe_up_gemm<<<dim3(T / 16, E, 8), 128, 0, stream>>>(downb, Wub, lists, gatew, counts, out);
}
